// MaskedMlp_20744692039972
// MI455X (gfx1250) — matrix operand order repaired and checked by reading, not compiled, not run on hardware
//
#include <hip/hip_runtime.h>
#include <hip/hip_bf16.h>

// MaskedMlp on gfx1250 (MI455X): per-class masked weights pre-packed to bf16,
// then two bf16-WMMA GEMMs (f32 accumulate) with LDS-staged activations and
// immediate-offset inner loops (no address math, no VALU in the K-loop).

typedef __attribute__((ext_vector_type(16))) __bf16 v16bf;
typedef __attribute__((ext_vector_type(8)))  __bf16 v8bf;
typedef __attribute__((ext_vector_type(4)))  __bf16 v4bf;
typedef __attribute__((ext_vector_type(8)))  float  v8f;

#define B_  16
#define N_  196
#define D_  768
#define H_  3072
#define C_  10
#define HD_ ((size_t)H_ * (size_t)D_)   // 2359296, same count for both layers

#define WG_THREADS 256
#define LDS1_STRIDE 776                 // 768 + 8 bf16 pad (bank-conflict free)

__device__ __forceinline__ float gelu_exact(float v) {
    return 0.5f * v * (1.0f + erff(v * 0.7071067811865475f));
}

__device__ __forceinline__ v16bf wmma_bf16(const v16bf& a, const v16bf& b, const v8f& c);

// =====================================================================
// Pack kernel: out[c][i] = bf16(W[i] * mask[c][i]) for all C classes.
// grid = (HD/1024, C), block = 256, 4 elements/thread, b64 stores.
// =====================================================================
__global__ __launch_bounds__(WG_THREADS)
void mask_pack_kernel(const float* __restrict__ W,
                      const float* __restrict__ mask,
                      __bf16*      __restrict__ out)
{
    const size_t i = ((size_t)blockIdx.x * WG_THREADS + threadIdx.x) * 4;
    const size_t ci = (size_t)blockIdx.y * HD_ + i;
    const float4 wv = *(const float4*)(W + i);
    const float4 mv = *(const float4*)(mask + ci);
    v4bf o;
    o[0] = (__bf16)(wv.x * mv.x);
    o[1] = (__bf16)(wv.y * mv.y);
    o[2] = (__bf16)(wv.z * mv.z);
    o[3] = (__bf16)(wv.w * mv.w);
    *(v4bf*)(out + ci) = o;
}

// =====================================================================
// fc1: h = GELU(x @ Wpk1[y[b]]^T + b1)  -> bf16 hbuf
// WG tile: 32 rows x 256 cols; wave tile: 32x32 (4 accumulators).
// =====================================================================
__global__ __launch_bounds__(WG_THREADS)
void fc1_pk_kernel(const float*  __restrict__ x,
                   const int*    __restrict__ y,
                   const __bf16* __restrict__ Wpk,
                   const float*  __restrict__ b1,
                   __bf16*       __restrict__ hbuf)
{
    const int b    = blockIdx.z;
    const int n0   = blockIdx.y * 32;
    const int c0   = blockIdx.x * 256;
    const int tid  = threadIdx.x;
    const int lane = tid & 31;
    const int wave = tid >> 5;

    __shared__ __bf16 lds_x[32 * LDS1_STRIDE];

    // Stage 32x768 x-tile (f32 -> bf16); 96 elems/thread, uniform trip count.
    for (int idx = tid; idx < 32 * D_; idx += WG_THREADS) {
        int r = idx / D_;
        int c = idx - r * D_;
        int n = n0 + r; if (n > N_ - 1) n = N_ - 1;
        lds_x[r * LDS1_STRIDE + c] = (__bf16)x[((size_t)b * N_ + n) * D_ + c];
    }
    __syncthreads();

    const int arow  = lane & 15;
    const int akoff = (lane >> 4) * 8;
    const int bcol  = lane & 15;
    const int bkoff = (lane >> 4) * 8;

    // Lane-fixed bases; all K-loop offsets are compile-time immediates.
    const __bf16* a0 = &lds_x[arow * LDS1_STRIDE + akoff];
    const __bf16* w0 = Wpk + (size_t)y[b] * HD_
                           + (size_t)(c0 + wave * 32 + bcol) * D_ + bkoff;

    v8f acc[4];
    #pragma unroll
    for (int t = 0; t < 4; ++t)
        #pragma unroll
        for (int i = 0; i < 8; ++i) acc[t][i] = 0.0f;

    #pragma unroll
    for (int it = 0; it < D_ / 32; ++it) {
        const int k0 = it * 32;
        v16bf A[2], Bv[2];
        #pragma unroll
        for (int m = 0; m < 2; ++m) {
            const v8bf lo = *(const v8bf*)(a0 + m * 16 * LDS1_STRIDE + k0);
            const v8bf hi = *(const v8bf*)(a0 + m * 16 * LDS1_STRIDE + k0 + 16);
            #pragma unroll
            for (int i = 0; i < 8; ++i) { A[m][i] = lo[i]; A[m][i + 8] = hi[i]; }
        }
        #pragma unroll
        for (int j = 0; j < 2; ++j) {
            const v8bf blo = *(const v8bf*)(w0 + j * 16 * D_ + k0);
            const v8bf bhi = *(const v8bf*)(w0 + j * 16 * D_ + k0 + 16);
            #pragma unroll
            for (int i = 0; i < 8; ++i) { Bv[j][i] = blo[i]; Bv[j][i + 8] = bhi[i]; }
        }
        #pragma unroll
        for (int m = 0; m < 2; ++m)
            #pragma unroll
            for (int j = 0; j < 2; ++j)
                acc[m * 2 + j] = __builtin_amdgcn_wmma_f32_16x16x32_bf16(
                    false, A[m], false, Bv[j], (short)0, acc[m * 2 + j], false, false);
    }

    const int mrow0 = (lane >> 4) * 8;
    #pragma unroll
    for (int j = 0; j < 2; ++j) {
        const int col  = c0 + wave * 32 + j * 16 + bcol;
        const float bias = b1[col];
        #pragma unroll
        for (int m = 0; m < 2; ++m)
            #pragma unroll
            for (int r = 0; r < 8; ++r) {
                const int n = n0 + m * 16 + mrow0 + r;
                if (n < N_) {
                    const float v = gelu_exact(acc[m * 2 + j][r] + bias);
                    hbuf[((size_t)b * N_ + n) * H_ + col] = (__bf16)v;
                }
            }
    }
}

// =====================================================================
// fc2: out = h @ Wpk2[y[b]]^T + b2  -> f32
// Same tiling; A read directly from bf16 hbuf (global), K = 3072.
// =====================================================================
__global__ __launch_bounds__(WG_THREADS)
void fc2_pk_kernel(const __bf16* __restrict__ hbuf,
                   const int*    __restrict__ y,
                   const __bf16* __restrict__ Wpk,
                   const float*  __restrict__ b2,
                   float*        __restrict__ out)
{
    const int b    = blockIdx.z;
    const int n0   = blockIdx.y * 32;
    const int c0   = blockIdx.x * 256;
    const int lane = threadIdx.x & 31;
    const int wave = threadIdx.x >> 5;

    const int arow  = lane & 15;
    const int akoff = (lane >> 4) * 8;
    const int bcol  = lane & 15;
    const int bkoff = (lane >> 4) * 8;

    int an0 = n0 + arow;      if (an0 > N_ - 1) an0 = N_ - 1;
    int an1 = n0 + 16 + arow; if (an1 > N_ - 1) an1 = N_ - 1;
    const __bf16* ap[2] = {
        hbuf + ((size_t)b * N_ + an0) * H_ + akoff,
        hbuf + ((size_t)b * N_ + an1) * H_ + akoff
    };
    const __bf16* w0 = Wpk + (size_t)y[b] * HD_
                           + (size_t)(c0 + wave * 32 + bcol) * H_ + bkoff;

    v8f acc[4];
    #pragma unroll
    for (int t = 0; t < 4; ++t)
        #pragma unroll
        for (int i = 0; i < 8; ++i) acc[t][i] = 0.0f;

    #pragma unroll 4
    for (int it = 0; it < H_ / 32; ++it) {
        const int k0 = it * 32;
        v16bf A[2], Bv[2];
        #pragma unroll
        for (int m = 0; m < 2; ++m) {
            const v8bf lo = *(const v8bf*)(ap[m] + k0);
            const v8bf hi = *(const v8bf*)(ap[m] + k0 + 16);
            #pragma unroll
            for (int i = 0; i < 8; ++i) { A[m][i] = lo[i]; A[m][i + 8] = hi[i]; }
        }
        #pragma unroll
        for (int j = 0; j < 2; ++j) {
            const v8bf blo = *(const v8bf*)(w0 + j * 16 * H_ + k0);
            const v8bf bhi = *(const v8bf*)(w0 + j * 16 * H_ + k0 + 16);
            #pragma unroll
            for (int i = 0; i < 8; ++i) { Bv[j][i] = blo[i]; Bv[j][i + 8] = bhi[i]; }
        }
        #pragma unroll
        for (int m = 0; m < 2; ++m)
            #pragma unroll
            for (int j = 0; j < 2; ++j)
                acc[m * 2 + j] = __builtin_amdgcn_wmma_f32_16x16x32_bf16(
                    false, A[m], false, Bv[j], (short)0, acc[m * 2 + j], false, false);
    }

    const int mrow0 = (lane >> 4) * 8;
    #pragma unroll
    for (int j = 0; j < 2; ++j) {
        const int col  = c0 + wave * 32 + j * 16 + bcol;
        const float bias = b2[col];
        #pragma unroll
        for (int m = 0; m < 2; ++m)
            #pragma unroll
            for (int r = 0; r < 8; ++r) {
                const int n = n0 + m * 16 + mrow0 + r;
                if (n < N_)
                    out[((size_t)b * N_ + n) * D_ + col] = acc[m * 2 + j][r] + bias;
            }
    }
}

// =====================================================================
// Fallback (small workspace): fused on-the-fly masking, 16x256 WG tile.
// =====================================================================
__global__ __launch_bounds__(WG_THREADS)
void fc1_fused_kernel(const float* __restrict__ x, const int* __restrict__ y,
                      const float* __restrict__ W1, const float* __restrict__ b1,
                      const float* __restrict__ mask1, __bf16* __restrict__ hbuf)
{
    const int b = blockIdx.z, n0 = blockIdx.y * 16, c0 = blockIdx.x * 256;
    const int tid = threadIdx.x, lane = tid & 31, wave = tid >> 5;
    __shared__ __bf16 lds_x[16 * LDS1_STRIDE];
    for (int idx = tid; idx < 16 * D_; idx += WG_THREADS) {
        int r = idx / D_, c = idx - r * D_;
        int n = n0 + r; if (n > N_ - 1) n = N_ - 1;
        lds_x[r * LDS1_STRIDE + c] = (__bf16)x[((size_t)b * N_ + n) * D_ + c];
    }
    __syncthreads();
    const float* mbase = mask1 + (size_t)y[b] * HD_;
    const int arow = lane & 15, akoff = (lane >> 4) * 8;
    const int bcol = lane & 15, bkoff = (lane >> 4) * 8;
    const int colbase = c0 + wave * 32;
    v8f acc[2];
    #pragma unroll
    for (int j = 0; j < 2; ++j)
        #pragma unroll
        for (int i = 0; i < 8; ++i) acc[j][i] = 0.0f;
    for (int k0 = 0; k0 < D_; k0 += 32) {
        v16bf afrag;
        const v8bf lo = *(const v8bf*)&lds_x[arow * LDS1_STRIDE + k0 + akoff];
        const v8bf hi = *(const v8bf*)&lds_x[arow * LDS1_STRIDE + k0 + akoff + 16];
        #pragma unroll
        for (int i = 0; i < 8; ++i) { afrag[i] = lo[i]; afrag[i + 8] = hi[i]; }
        #pragma unroll
        for (int j = 0; j < 2; ++j) {
            const int hcol = colbase + j * 16 + bcol;
            const float* wp = W1    + (size_t)hcol * D_ + k0 + bkoff;
            const float* mp = mbase + (size_t)hcol * D_ + k0 + bkoff;
            v16bf bfrag;
            #pragma unroll
            for (int i = 0; i < 16; i += 4) {
                const float4 wv = *(const float4*)(wp + i + (i >= 8 ? 8 : 0));
                const float4 mv = *(const float4*)(mp + i + (i >= 8 ? 8 : 0));
                bfrag[i + 0] = (__bf16)(wv.x * mv.x);
                bfrag[i + 1] = (__bf16)(wv.y * mv.y);
                bfrag[i + 2] = (__bf16)(wv.z * mv.z);
                bfrag[i + 3] = (__bf16)(wv.w * mv.w);
            }
            acc[j] = __builtin_amdgcn_wmma_f32_16x16x32_bf16(
                false, afrag, false, bfrag, (short)0, acc[j], false, false);
        }
    }
    const int mrow0 = (lane >> 4) * 8;
    #pragma unroll
    for (int j = 0; j < 2; ++j) {
        const int hcol = colbase + j * 16 + bcol;
        const float bias = b1[hcol];
        #pragma unroll
        for (int r = 0; r < 8; ++r) {
            const int n = n0 + mrow0 + r;
            if (n < N_)
                hbuf[((size_t)b * N_ + n) * H_ + hcol] =
                    (__bf16)gelu_exact(acc[j][r] + bias);
        }
    }
}

__global__ __launch_bounds__(WG_THREADS)
void fc2_fused_kernel(const __bf16* __restrict__ hbuf, const int* __restrict__ y,
                      const float* __restrict__ W2, const float* __restrict__ b2,
                      const float* __restrict__ mask2, float* __restrict__ out)
{
    const int b = blockIdx.z, n0 = blockIdx.y * 16, c0 = blockIdx.x * 256;
    const int lane = threadIdx.x & 31, wave = threadIdx.x >> 5;
    const float* mbase = mask2 + (size_t)y[b] * HD_;
    const int arow = lane & 15, akoff = (lane >> 4) * 8;
    const int bcol = lane & 15, bkoff = (lane >> 4) * 8;
    const int colbase = c0 + wave * 32;
    int an = n0 + arow; if (an > N_ - 1) an = N_ - 1;
    const __bf16* arowp = hbuf + ((size_t)b * N_ + an) * H_;
    v8f acc[2];
    #pragma unroll
    for (int j = 0; j < 2; ++j)
        #pragma unroll
        for (int i = 0; i < 8; ++i) acc[j][i] = 0.0f;
    for (int k0 = 0; k0 < H_; k0 += 32) {
        v16bf afrag;
        const v8bf lo = *(const v8bf*)(arowp + k0 + akoff);
        const v8bf hi = *(const v8bf*)(arowp + k0 + akoff + 16);
        #pragma unroll
        for (int i = 0; i < 8; ++i) { afrag[i] = lo[i]; afrag[i + 8] = hi[i]; }
        #pragma unroll
        for (int j = 0; j < 2; ++j) {
            const int dcol = colbase + j * 16 + bcol;
            const float* wp = W2    + (size_t)dcol * H_ + k0 + bkoff;
            const float* mp = mbase + (size_t)dcol * H_ + k0 + bkoff;
            v16bf bfrag;
            #pragma unroll
            for (int i = 0; i < 16; i += 4) {
                const float4 wv = *(const float4*)(wp + i + (i >= 8 ? 8 : 0));
                const float4 mv = *(const float4*)(mp + i + (i >= 8 ? 8 : 0));
                bfrag[i + 0] = (__bf16)(wv.x * mv.x);
                bfrag[i + 1] = (__bf16)(wv.y * mv.y);
                bfrag[i + 2] = (__bf16)(wv.z * mv.z);
                bfrag[i + 3] = (__bf16)(wv.w * mv.w);
            }
            acc[j] = __builtin_amdgcn_wmma_f32_16x16x32_bf16(
                false, afrag, false, bfrag, (short)0, acc[j], false, false);
        }
    }
    const int mrow0 = (lane >> 4) * 8;
    #pragma unroll
    for (int j = 0; j < 2; ++j) {
        const int dcol = colbase + j * 16 + bcol;
        const float bias = b2[dcol];
        #pragma unroll
        for (int r = 0; r < 8; ++r) {
            const int n = n0 + mrow0 + r;
            if (n < N_)
                out[((size_t)b * N_ + n) * D_ + dcol] = acc[j][r] + bias;
        }
    }
}

// =====================================================================
extern "C" void kernel_launch(void* const* d_in, const int* in_sizes, int n_in,
                              void* d_out, int out_size, void* d_ws, size_t ws_size,
                              hipStream_t stream) {
    (void)in_sizes; (void)n_in; (void)out_size;
    const float* x     = (const float*)d_in[0];
    const int*   y     = (const int*)  d_in[1];
    const float* W1    = (const float*)d_in[2];
    const float* b1    = (const float*)d_in[3];
    const float* W2    = (const float*)d_in[4];
    const float* b2    = (const float*)d_in[5];
    const float* mask1 = (const float*)d_in[6];
    const float* mask2 = (const float*)d_in[7];
    float* out = (float*)d_out;

    const size_t slab_bytes = (size_t)C_ * HD_ * sizeof(__bf16);        // 47.2 MB
    const size_t h_bytes    = (size_t)B_ * N_ * H_ * sizeof(__bf16);    // 18.4 MB
    dim3 block(WG_THREADS);

    if (ws_size >= slab_bytes + h_bytes) {
        __bf16* wslab = (__bf16*)d_ws;                       // reused by both layers
        __bf16* hbuf  = (__bf16*)((char*)d_ws + slab_bytes);

        dim3 gpack(HD_ / (WG_THREADS * 4), C_, 1);           // 2304 x 10
        dim3 g1(H_ / 256, (N_ + 31) / 32, B_);               // 12 x 7 x 16
        dim3 g2(D_ / 256, (N_ + 31) / 32, B_);               //  3 x 7 x 16

        mask_pack_kernel<<<gpack, block, 0, stream>>>(W1, mask1, wslab);
        fc1_pk_kernel   <<<g1,   block, 0, stream>>>(x, y, wslab, b1, hbuf);
        mask_pack_kernel<<<gpack, block, 0, stream>>>(W2, mask2, wslab); // overwrite
        fc2_pk_kernel   <<<g2,   block, 0, stream>>>(hbuf, y, wslab, b2, out);
    } else {
        __bf16* hbuf = (__bf16*)d_ws;
        dim3 g1(H_ / 256, (N_ + 15) / 16, B_);               // 12 x 13 x 16
        dim3 g2(D_ / 256, (N_ + 15) / 16, B_);               //  3 x 13 x 16
        fc1_fused_kernel<<<g1, block, 0, stream>>>(x, y, W1, b1, mask1, hbuf);
        fc2_fused_kernel<<<g2, block, 0, stream>>>(hbuf, y, W2, b2, mask2, out);
    }
}
